// MultiHeadSelfAttention_70918499991560
// MI455X (gfx1250) — matrix operand order repaired and checked by reading, not compiled, not run on hardware
//
#include <hip/hip_runtime.h>

// ---------------------------------------------------------------------------
// MI455X (gfx1250, wave32) multi-head self-attention forward.
// All matrix math via v_wmma_f32_16x16x32_bf16 (fp32 accumulate).
//
// Pipeline:
//   cvt fp32->bf16 | QKV proj (LDS-staged WMMA GEMM, 64x64 block tile,
//   async global->LDS A staging when available)
//   | rowstats: online softmax max/sum via WMMA logits recompute (no attn IO)
//   | attn+ctx fused: recompute logits, normalize, single NT write of attn,
//     bf16 tile -> LDS -> ctx WMMA against LDS-transposed V tile
//   | out proj (same GEMM, fp32 into d_out)
//
// Roofline: attn (1.07 GB) is written exactly once; everything else is
// L2-resident (192 MB L2 >> weights+activations). Math is bf16 WMMA.
// ---------------------------------------------------------------------------

typedef __bf16 bf16;
typedef __attribute__((ext_vector_type(16))) __bf16 v16bf;
typedef __attribute__((ext_vector_type(8)))  float  v8f;
typedef __attribute__((ext_vector_type(4)))  float  f32x4;
typedef __attribute__((ext_vector_type(2)))  float  f32x2;
typedef __attribute__((ext_vector_type(4)))  unsigned int u32x4;
typedef __attribute__((ext_vector_type(2)))  unsigned int u32x2;
typedef __attribute__((ext_vector_type(4)))  int v4i;

union FragU {
  v16bf v;
  u32x4 q[2];
  bf16  b[16];
};

#define NB 4
#define NS 2048
#define NE 1024
#define NH 16
#define ND 64
#define NM (NB * NS)

// CDNA5 async global->LDS copy path (ASYNCcnt), guarded so compile never fails
#if __has_builtin(__builtin_amdgcn_global_load_async_to_lds_b128) && \
    __has_builtin(__builtin_amdgcn_s_wait_asynccnt)
#define USE_ASYNC_COPY 1
#else
#define USE_ASYNC_COPY 0
#endif

#define AS1 __attribute__((address_space(1)))
#define AS3 __attribute__((address_space(3)))

__device__ __forceinline__ v8f vzero8() {
  v8f z;
#pragma unroll
  for (int i = 0; i < 8; ++i) z[i] = 0.0f;
  return z;
}

__device__ __forceinline__ v8f wmma_bf16(const FragU& a, const FragU& b, v8f c) {
  return __builtin_amdgcn_wmma_f32_16x16x32_bf16(
      false, a.v, false, b.v, (short)0, c, false, false);
}

// ---------------------------------------------------------------------------
// fp32 -> bf16, 4 elements / thread (b128 load, b64 store)
// ---------------------------------------------------------------------------
__global__ __launch_bounds__(256) void cvt_f32_to_bf16_x4(
    const f32x4* __restrict__ in, u32x2* __restrict__ out, int n4)
{
  int i = blockIdx.x * 256 + threadIdx.x;
  if (i >= n4) return;
  f32x4 v = in[i];
  union { bf16 h[4]; u32x2 u; } p;
  p.h[0] = (bf16)v[0]; p.h[1] = (bf16)v[1];
  p.h[2] = (bf16)v[2]; p.h[3] = (bf16)v[3];
  out[i] = p.u;
}

// ---------------------------------------------------------------------------
// LDS-staged GEMM: C[M,N] = A[M,K] * W[K,N] + bias.
// Block = 256 thr = 8 waves, block tile 64(M) x 64(N), k-step 32.
// Wave (r,c): r=w>>2 -> 32-row M half, c=w&3 -> 16-col N quarter.
// Each wave computes TWO 16x16 tiles sharing one B fragment (2 WMMA/barrier).
// A tile staged with GLOBAL_LOAD_ASYNC_TO_LDS_B128 when available.
// LDS row stride 40 halves (80 B) keeps every fragment read 16B-aligned.
// TRANS_HEADS: write output as [B,H,S,D] bf16. OUT_F32: fp32 row-major.
// ---------------------------------------------------------------------------
template <int TRANS_HEADS, int OUT_F32>
__global__ __launch_bounds__(256) void gemm_bias_kernel(
    const bf16* __restrict__ A, const bf16* __restrict__ W,
    const float* __restrict__ bias,
    bf16* __restrict__ outB, float* __restrict__ outF,
    int Mtot, int Ktot, int Ntot)
{
  __shared__ bf16 As[64][40];   // A tile  [m][k]
  __shared__ bf16 Bs[64][40];   // W tile transposed: [n][k]

  const int t    = threadIdx.x;
  const int m0   = blockIdx.y * 64;
  const int n0   = blockIdx.x * 64;
  const int w    = t >> 5, lane = t & 31;
  const int r    = w >> 2, c    = w & 3;
  const int g    = lane >> 4, l15 = lane & 15;

  v8f acc0 = vzero8(), acc1 = vzero8();

  for (int k0 = 0; k0 < Ktot; k0 += 32) {
    __syncthreads();
    { // stage A: 64 rows x 32 cols bf16, 16B per thread (async when possible)
      int row = t >> 2, col = (t & 3) * 8;
      const bf16* src = A + (size_t)(m0 + row) * Ktot + k0 + col;
#if USE_ASYNC_COPY
      __builtin_amdgcn_global_load_async_to_lds_b128(
          (AS1 v4i*)src, (AS3 v4i*)&As[row][col], 0, 0);
#else
      u32x4 v = *(const u32x4*)src;
      *(u32x4*)&As[row][col] = v;
#endif
    }
    { // stage W transposed: 32(k) x 64(n), 8 elems (16B) per thread
      int kk = t >> 3, nn = (t & 7) * 8;
      u32x4 v = *(const u32x4*)(W + (size_t)(k0 + kk) * Ntot + n0 + nn);
      const bf16* hb = (const bf16*)&v;
#pragma unroll
      for (int j = 0; j < 8; ++j) Bs[nn + j][kk] = hb[j];
    }
#if USE_ASYNC_COPY
    __builtin_amdgcn_s_wait_asynccnt(0);
#endif
    __syncthreads();

    FragU af0, af1, bw;
    int row0 = r * 32 + l15;
    af0.q[0] = *(const u32x4*)&As[row0][g * 8];
    af0.q[1] = *(const u32x4*)&As[row0][16 + g * 8];
    af1.q[0] = *(const u32x4*)&As[row0 + 16][g * 8];
    af1.q[1] = *(const u32x4*)&As[row0 + 16][16 + g * 8];
    int nrow = c * 16 + l15;
    bw.q[0] = *(const u32x4*)&Bs[nrow][g * 8];
    bw.q[1] = *(const u32x4*)&Bs[nrow][16 + g * 8];
    acc0 = wmma_bf16(af0, bw, acc0);
    acc1 = wmma_bf16(af1, bw, acc1);
  }

  // C/D layout: lane col N = l15; VGPR i -> row i + 8*g
  const int col = n0 + c * 16 + l15;
  const float bv = bias[col];
#pragma unroll
  for (int i = 0; i < 8; ++i) {
#pragma unroll
    for (int half = 0; half < 2; ++half) {
      int m = m0 + r * 32 + half * 16 + g * 8 + i;
      float val = (half ? acc1[i] : acc0[i]) + bv;
      if (TRANS_HEADS) {
        int b = m >> 11, s = m & (NS - 1);      // m = b*S + s
        int h = col >> 6, d = col & (ND - 1);   // col = h*D + d
        outB[(((size_t)(b * NH + h)) * NS + s) * ND + d] = (bf16)val;
      } else if (OUT_F32) {
        outF[(size_t)m * Ntot + col] = val;
      } else {
        outB[(size_t)m * Ntot + col] = (bf16)val;
      }
    }
  }
}

// ---------------------------------------------------------------------------
// Pass 1: per (bh, 32-query tile) compute row max & sum(exp) of
// logits = scale*Q.K^T + mask*-1e9 with ONLINE softmax stats — no attn IO.
// Wave w: r=w>>2 -> query sub-tile, c=w&3 -> key stripe (keys (4j+c)*16).
// Per-lane (m,s) for 8 rows; reduce over 16 lanes (shfl_xor), then over the
// 4 key-striped waves via LDS. Output stats[bh*S+q] = (max, 1/sum).
// ---------------------------------------------------------------------------
__global__ __launch_bounds__(256) void rowstats_kernel(
    const bf16* __restrict__ qh, const bf16* __restrict__ kh,
    const int* __restrict__ mask, f32x2* __restrict__ stats)
{
  __shared__ float redM[32][4];
  __shared__ float redS[32][4];

  const int t = threadIdx.x, w = t >> 5, lane = t & 31;
  const int r = w >> 2, c = w & 3;
  const int g = lane >> 4, l15 = lane & 15;
  const int bh = blockIdx.x;
  const int b  = bh >> 4;
  const int q0 = blockIdx.y * 32 + r * 16;
  const bf16* Qp = qh + (size_t)bh * NS * ND;
  const bf16* Kp = kh + (size_t)bh * NS * ND;

  // Q fragments for rows q0.. (fixed all loop): d0 = 0 and 32
  FragU aq0, aq1;
  {
    const bf16* qptr = Qp + (size_t)(q0 + l15) * ND + g * 8;
    aq0.q[0] = *(const u32x4*)qptr;
    aq0.q[1] = *(const u32x4*)(qptr + 16);
    aq1.q[0] = *(const u32x4*)(qptr + 32);
    aq1.q[1] = *(const u32x4*)(qptr + 48);
  }

  float m[8], s[8];
#pragma unroll
  for (int i = 0; i < 8; ++i) { m[i] = -3.0e38f; s[i] = 0.0f; }

  for (int j = 0; j < NS / 64; ++j) {     // 32 iterations, 64 keys per block
    const int kt = (j * 4 + c) * 16;
    v8f acc = vzero8();
    FragU bk;
    const bf16* kptr = Kp + (size_t)(kt + l15) * ND + g * 8;
    bk.q[0] = *(const u32x4*)kptr;
    bk.q[1] = *(const u32x4*)(kptr + 16);
    acc = wmma_bf16(aq0, bk, acc);
    bk.q[0] = *(const u32x4*)(kptr + 32);
    bk.q[1] = *(const u32x4*)(kptr + 48);
    acc = wmma_bf16(aq1, bk, acc);

    const int key = kt + l15;
    const float madd = (float)mask[b * NS + key] * -1e9f;
#pragma unroll
    for (int i = 0; i < 8; ++i) {
      float x = acc[i] * 0.125f + madd;
      float mn = fmaxf(m[i], x);
      s[i] = s[i] * __expf(m[i] - mn) + __expf(x - mn);
      m[i] = mn;
    }
  }

  // reduce (m,s) across the 16 lanes of each half-wave (keys differ by lane)
#pragma unroll
  for (int d = 1; d < 16; d <<= 1) {
#pragma unroll
    for (int i = 0; i < 8; ++i) {
      float m2 = __shfl_xor(m[i], d, 32);
      float s2 = __shfl_xor(s[i], d, 32);
      float mn = fmaxf(m[i], m2);
      s[i] = s[i] * __expf(m[i] - mn) + s2 * __expf(m2 - mn);
      m[i] = mn;
    }
  }

  if (l15 == 0) {
#pragma unroll
    for (int i = 0; i < 8; ++i) {
      int rr = r * 16 + g * 8 + i;   // 0..31 (row within block)
      redM[rr][c] = m[i];
      redS[rr][c] = s[i];
    }
  }
  __syncthreads();
  if (t < 32) {
    float mm = redM[t][0], ss = redS[t][0];
#pragma unroll
    for (int cc = 1; cc < 4; ++cc) {
      float m2 = redM[t][cc], s2 = redS[t][cc];
      float mn = fmaxf(mm, m2);
      ss = ss * __expf(mm - mn) + s2 * __expf(m2 - mn);
      mm = mn;
    }
    int qrow = blockIdx.y * 32 + t;
    f32x2 st; st[0] = mm; st[1] = 1.0f / ss;
    stats[(size_t)bh * NS + qrow] = st;
  }
}

// ---------------------------------------------------------------------------
// Pass 2: per (bh, 32-query tile), loop keys in 32-chunks:
//   waves 0..3 recompute the 32x32 logits tile (WMMA), normalize with the
//   precomputed stats, NT-store attn (its ONLY HBM write) and drop bf16 into
//   LDS; all 8 waves then run ctx WMMA against the LDS-transposed V tile.
// ctx written bf16 in [B,S,E] for the final projection.
// ---------------------------------------------------------------------------
__global__ __launch_bounds__(256) void attn_ctx_kernel(
    const bf16* __restrict__ qh, const bf16* __restrict__ kh,
    const bf16* __restrict__ vh, const int* __restrict__ mask,
    const f32x2* __restrict__ stats, float* __restrict__ attn,
    bf16* __restrict__ ctx)
{
  __shared__ bf16 Asb[32][40];  // normalized attn tile, bf16 [q][k]
  __shared__ bf16 Vs[64][40];   // V tile transposed: [d][k]

  const int t = threadIdx.x, w = t >> 5, lane = t & 31;
  const int g = lane >> 4, l15 = lane & 15;
  const int bh = blockIdx.x, b = bh >> 4;
  const int q0 = blockIdx.y * 32;
  const bf16* Qp = qh + (size_t)bh * NS * ND;
  const bf16* Kp = kh + (size_t)bh * NS * ND;
  const bf16* Vb = vh + (size_t)bh * NS * ND;
  float* aout = attn + (size_t)bh * NS * NS;

  // ctx mapping (all 8 waves): cr -> query sub-tile, cc -> d column sub-tile
  const int cr = w >> 2, cc = w & 3;
  v8f ctxacc = vzero8();

  // logits mapping (waves 0..3): lr -> query sub-tile, lk -> key sub-tile
  const bool isLog = (w < 4);
  const int lr = (w >> 1) & 1, lk = w & 1;

  FragU aq0, aq1;
  float lm[8], linv[8];
  if (isLog) {
    const bf16* qptr = Qp + (size_t)(q0 + lr * 16 + l15) * ND + g * 8;
    aq0.q[0] = *(const u32x4*)qptr;
    aq0.q[1] = *(const u32x4*)(qptr + 16);
    aq1.q[0] = *(const u32x4*)(qptr + 32);
    aq1.q[1] = *(const u32x4*)(qptr + 48);
#pragma unroll
    for (int i = 0; i < 8; ++i) {
      int qrow = q0 + lr * 16 + g * 8 + i;
      f32x2 st = stats[(size_t)bh * NS + qrow];
      lm[i] = st[0]; linv[i] = st[1];
    }
  }

  for (int k0 = 0; k0 < NS; k0 += 32) {
    __syncthreads();            // protect previous iteration's LDS reads
    { // stage V transposed: 32(k) x 64(d), 16B per thread
      int kk = t >> 3, dd = (t & 7) * 8;
      u32x4 v = *(const u32x4*)(Vb + (size_t)(k0 + kk) * ND + dd);
      const bf16* hb = (const bf16*)&v;
#pragma unroll
      for (int j = 0; j < 8; ++j) Vs[dd + j][kk] = hb[j];
    }
    if (isLog) {
      const int kt = k0 + lk * 16;
      v8f acc = vzero8();
      FragU bk;
      const bf16* kptr = Kp + (size_t)(kt + l15) * ND + g * 8;
      bk.q[0] = *(const u32x4*)kptr;
      bk.q[1] = *(const u32x4*)(kptr + 16);
      acc = wmma_bf16(aq0, bk, acc);
      bk.q[0] = *(const u32x4*)(kptr + 32);
      bk.q[1] = *(const u32x4*)(kptr + 48);
      acc = wmma_bf16(aq1, bk, acc);

      const int key = kt + l15;
      const float madd = (float)mask[b * NS + key] * -1e9f;
#pragma unroll
      for (int i = 0; i < 8; ++i) {
        float x = acc[i] * 0.125f + madd;
        float p = __expf(x - lm[i]) * linv[i];
        int qrow = q0 + lr * 16 + g * 8 + i;
        __builtin_nontemporal_store(p, &aout[(size_t)qrow * NS + key]);
        Asb[lr * 16 + g * 8 + i][lk * 16 + l15] = (bf16)p;
      }
    }
    __syncthreads();

    // ctx WMMA (all 8 waves)
    FragU aa, bb;
    int arow = cr * 16 + l15;
    aa.q[0] = *(const u32x4*)&Asb[arow][g * 8];
    aa.q[1] = *(const u32x4*)&Asb[arow][16 + g * 8];
    int drow = cc * 16 + l15;
    bb.q[0] = *(const u32x4*)&Vs[drow][g * 8];
    bb.q[1] = *(const u32x4*)&Vs[drow][16 + g * 8];
    ctxacc = wmma_bf16(aa, bb, ctxacc);
  }

  const int h = bh & 15;
  const int dcol = cc * 16 + l15;
#pragma unroll
  for (int i = 0; i < 8; ++i) {
    int sI = q0 + cr * 16 + g * 8 + i;
    ctx[((size_t)(b * NS + sI)) * NE + h * ND + dcol] = (bf16)ctxacc[i];
  }
}

// ---------------------------------------------------------------------------
// Host launcher
// ---------------------------------------------------------------------------
extern "C" void kernel_launch(void* const* d_in, const int* in_sizes, int n_in,
                              void* d_out, int out_size, void* d_ws, size_t ws_size,
                              hipStream_t stream)
{
  (void)in_sizes; (void)n_in; (void)out_size; (void)ws_size;

  const float* v_in = (const float*)d_in[0];
  const float* k_in = (const float*)d_in[1];
  const float* q_in = (const float*)d_in[2];
  const int*   mask = (const int*)d_in[3];
  const float* wq = (const float*)d_in[4];
  const float* bq = (const float*)d_in[5];
  const float* wk = (const float*)d_in[6];
  const float* bk = (const float*)d_in[7];
  const float* wv = (const float*)d_in[8];
  const float* bv = (const float*)d_in[9];
  const float* wo = (const float*)d_in[10];
  const float* bo = (const float*)d_in[11];

  float* out  = (float*)d_out;
  float* attn = out + (size_t)NB * NS * NE;   // [B,H,S,S] region of d_out

  const size_t BSE = (size_t)NB * NS * NE;    // 8,388,608
  const size_t EE  = (size_t)NE * NE;         // 1,048,576

  bf16* ws  = (bf16*)d_ws;
  bf16* qbf = ws;            ws += BSE;       // q as bf16 (reused as ctx)
  bf16* kbf = ws;            ws += BSE;
  bf16* vbf = ws;            ws += BSE;
  bf16* qhb = ws;            ws += BSE;       // [B,H,S,D]
  bf16* khb = ws;            ws += BSE;
  bf16* vhb = ws;            ws += BSE;
  bf16* wqb = ws;            ws += EE;
  bf16* wkb = ws;            ws += EE;
  bf16* wvb = ws;            ws += EE;
  bf16* wob = ws;            ws += EE;
  f32x2* stats = (f32x2*)ws;                  // [B*H*S] (max, 1/sum) ~1 MB

  // 1) fp32 -> bf16 conversions
  {
    int n4a = (int)(BSE / 4), n4w = (int)(EE / 4);
    cvt_f32_to_bf16_x4<<<n4a / 256, 256, 0, stream>>>((const f32x4*)q_in, (u32x2*)qbf, n4a);
    cvt_f32_to_bf16_x4<<<n4a / 256, 256, 0, stream>>>((const f32x4*)k_in, (u32x2*)kbf, n4a);
    cvt_f32_to_bf16_x4<<<n4a / 256, 256, 0, stream>>>((const f32x4*)v_in, (u32x2*)vbf, n4a);
    cvt_f32_to_bf16_x4<<<n4w / 256, 256, 0, stream>>>((const f32x4*)wq, (u32x2*)wqb, n4w);
    cvt_f32_to_bf16_x4<<<n4w / 256, 256, 0, stream>>>((const f32x4*)wk, (u32x2*)wkb, n4w);
    cvt_f32_to_bf16_x4<<<n4w / 256, 256, 0, stream>>>((const f32x4*)wv, (u32x2*)wvb, n4w);
    cvt_f32_to_bf16_x4<<<n4w / 256, 256, 0, stream>>>((const f32x4*)wo, (u32x2*)wob, n4w);
  }

  // 2) QKV projections -> head-split bf16 [B,H,S,D]
  dim3 gGemm(NE / 64, NM / 64);
  gemm_bias_kernel<1, 0><<<gGemm, 256, 0, stream>>>(qbf, wqb, bq, qhb, nullptr, NM, NE, NE);
  gemm_bias_kernel<1, 0><<<gGemm, 256, 0, stream>>>(kbf, wkb, bk, khb, nullptr, NM, NE, NE);
  gemm_bias_kernel<1, 0><<<gGemm, 256, 0, stream>>>(vbf, wvb, bv, vhb, nullptr, NM, NE, NE);

  // 3) Pass 1: online softmax row stats (no attn traffic)
  rowstats_kernel<<<dim3(NB * NH, NS / 32), 256, 0, stream>>>(qhb, khb, mask, stats);

  // 4) Pass 2: normalized attn (single NT write) + ctx, fused
  bf16* ctxb = qbf;
  attn_ctx_kernel<<<dim3(NB * NH, NS / 32), 256, 0, stream>>>(
      qhb, khb, vhb, mask, stats, attn, ctxb);

  // 5) out = ctx @ Wo + bo  (fp32 into d_out)
  gemm_bias_kernel<0, 1><<<gGemm, 256, 0, stream>>>(ctxb, wob, bo, nullptr, out, NM, NE, NE);
}
